// Self_attention_68221260529819
// MI455X (gfx1250) — compile-verified, matrix operand order checked by reading, not run on hardware
//
#include <hip/hip_runtime.h>
#include <hip/hip_bf16.h>

typedef __bf16 bf16_t;
typedef __attribute__((ext_vector_type(16))) __bf16 v16bf;
typedef __attribute__((ext_vector_type(8)))  float  v8f;

#define STRIDE 72                     // bf16 elems per LDS row (144 B, 16B-aligned rows)
#define BUF_ELEMS (64 * STRIDE)       // 4608 bf16 per 64x64 buffer
#define REGION_ELEMS (4 * BUF_ELEMS)  // 18432 bf16 (36864 B) per wave/channel
#define REGION_BYTES (REGION_ELEMS * 2)
#define OUT_BYTE_OFF (2 * BUF_ELEMS * 2)  // f32 out staging overlays Qb+Kb (18432 B)
#define OSTRIDE 68                    // f32 elems per staging row (272 B)

__device__ __forceinline__ v16bf load_frag(const bf16_t* base, int tileRow, int k0, int lane) {
    // 16-bit A/B fragment: lanes 0-15 -> rows M=0..15 (K 0..7,16..23),
    // lanes 16-31 -> same rows (K 8..15,24..31). Two contiguous b128 loads.
    const int r  = tileRow + (lane & 15);
    const int kh = (lane >> 4) << 3;           // 0 or 8
    const bf16_t* p = base + r * STRIDE + k0 + kh;
    union { v16bf v; struct { uint4 lo, hi; } q; } f;
    f.q.lo = *reinterpret_cast<const uint4*>(p);
    f.q.hi = *reinterpret_cast<const uint4*>(p + 16);
    return f.v;
}

__device__ __forceinline__ v8f wmma_bf16(v16bf a, v16bf b, v8f c) {
    return __builtin_amdgcn_wmma_f32_16x16x32_bf16(false, a, false, b, (short)0, c, false, false);
}

__device__ __forceinline__ v8f gemm_tile64(const bf16_t* A, const bf16_t* B,
                                           int tm, int tn, int lane) {
    v8f acc = {0.f, 0.f, 0.f, 0.f, 0.f, 0.f, 0.f, 0.f};
    acc = wmma_bf16(load_frag(A, tm,  0, lane), load_frag(B, tn,  0, lane), acc);
    acc = wmma_bf16(load_frag(A, tm, 32, lane), load_frag(B, tn, 32, lane), acc);
    return acc;
}

__device__ __forceinline__ float hsig(float x) {
    return fminf(fmaxf(fmaf(0.2f, x, 0.5f), 0.0f), 1.0f);
}

__device__ __forceinline__ void store_tile_bf16(bf16_t* dst, int tm, int tn, int lane,
                                                const v8f& val) {
    // C/D layout: VGPR v, lanes 0-15 -> M = v, N = lane; lanes 16-31 -> M = v+8, N = lane-16
    const int col  = tn + (lane & 15);
    const int row0 = tm + ((lane >> 4) << 3);
#pragma unroll
    for (int v = 0; v < 8; ++v)
        dst[(row0 + v) * STRIDE + col] = (bf16_t)val[v];
}

__global__ void self_attn_bf16_wmma(const float* __restrict__ x,
                                    const float* __restrict__ Wq,
                                    const float* __restrict__ Wk,
                                    float* __restrict__ out) {
    extern __shared__ __align__(16) char smem_raw[];
    bf16_t* smem = reinterpret_cast<bf16_t*>(smem_raw);

    const int blk  = blockIdx.x;       // 2048 = 32 b * 64 channel-groups
    const int b    = blk >> 6;
    const int c0   = (blk & 63) * 4;
    const int tid  = threadIdx.x;      // 128 threads = 4 waves (wave32)
    const int wave = tid >> 5;
    const int lane = tid & 31;
    const int c    = c0 + wave;        // this wave's channel

    bf16_t* Xb = smem + wave * REGION_ELEMS;   // X   (bf16, row-major [h][w])
    bf16_t* Wb = Xb + BUF_ELEMS;               // WqT / WkT / P^T (reused)
    bf16_t* Qb = Xb + 2 * BUF_ELEMS;           // q   (row-major [h][k])
    bf16_t* Kb = Xb + 3 * BUF_ELEMS;           // K   (row-major [m][k])
    bf16_t* Pt = Wb;                           // P^T ([m][w]) reuses W buffer
    // f32 output staging overlays Qb+Kb (dead after GEMM3); +wave*32 B bank skew
    float*  Ob = reinterpret_cast<float*>(smem_raw + wave * REGION_BYTES
                                          + OUT_BYTE_OFF + wave * 32);

    // ---- cooperative, coalesced load of X for 4 channels (transpose c->slab) ----
    {
        const float* xb = x + (size_t)b * 64 * 64 * 256 + c0;
        const int co = tid & 3;                // channel within group (contig 16 B)
        const int p0 = tid >> 2;               // 0..31
        bf16_t* Xc = smem + co * REGION_ELEMS; // Xb of wave 'co'
        for (int hw = p0; hw < 4096; hw += 32) {
            float v = xb[(size_t)hw * 256 + co];
            Xc[(hw >> 6) * STRIDE + (hw & 63)] = (bf16_t)v;
        }
    }
    // ---- per-wave: WqT into own region (coalesced read, transposed LDS store) ----
    {
        const float* wq = Wq + (size_t)c * 4096;
        for (int i = lane; i < 4096; i += 32) {
            const int w = i >> 6, k = i & 63;
            Wb[k * STRIDE + w] = (bf16_t)wq[i];
        }
    }
    __syncthreads();   // X slabs are written cross-wave

    // ================= GEMM1: q = hard_sigmoid(X @ Wq) =================
#pragma unroll
    for (int tm = 0; tm < 4; ++tm)
#pragma unroll
        for (int tn = 0; tn < 4; ++tn) {
            v8f acc = gemm_tile64(Xb, Wb, tm * 16, tn * 16, lane);
            v8f r;
#pragma unroll
            for (int v = 0; v < 8; ++v) r[v] = hsig(acc[v]);
            store_tile_bf16(Qb, tm * 16, tn * 16, lane, r);
        }

    // ---- WkT overwrites Wb (same-wave LDS ops are in-order) ----
    {
        const float* wk = Wk + (size_t)c * 4096;
        for (int i = lane; i < 4096; i += 32) {
            const int w = i >> 6, k = i & 63;
            Wb[k * STRIDE + w] = (bf16_t)wk[i];
        }
    }

    // ================= GEMM2: K = tanh(X @ Wk) =================
#pragma unroll
    for (int tm = 0; tm < 4; ++tm)
#pragma unroll
        for (int tn = 0; tn < 4; ++tn) {
            v8f acc = gemm_tile64(Xb, Wb, tm * 16, tn * 16, lane);
            v8f r;
#pragma unroll
            for (int v = 0; v < 8; ++v) {
                float e2 = __expf(2.0f * acc[v]);                   // tanh via exp
                r[v] = (e2 - 1.0f) * __builtin_amdgcn_rcpf(e2 + 1.0f);
            }
            store_tile_bf16(Kb, tm * 16, tn * 16, lane, r);
        }

    // ======== GEMM3 + softmax: P = softmax(hard_sigmoid(q @ K^T)), store P^T ========
    // Row-major Kb is exactly the B-fragment layout for q @ K^T.
#pragma unroll
    for (int tm = 0; tm < 4; ++tm) {
        v8f e[4];
        v8f psum = {0.f, 0.f, 0.f, 0.f, 0.f, 0.f, 0.f, 0.f};
#pragma unroll
        for (int tn = 0; tn < 4; ++tn) {
            v8f acc = gemm_tile64(Qb, Kb, tm * 16, tn * 16, lane);
#pragma unroll
            for (int v = 0; v < 8; ++v) {
                float s  = hsig(acc[v]);        // in [0,1] -> exp safe without max-sub
                float ev = __expf(s);
                e[tn][v] = ev;
                psum[v] += ev;
            }
        }
        // row sums: reduce across the 16 lanes of each half-wave (C-layout rows)
#pragma unroll
        for (int m = 1; m <= 8; m <<= 1)
#pragma unroll
            for (int v = 0; v < 8; ++v) psum[v] += __shfl_xor(psum[v], m, 32);

        const int row0 = tm * 16 + ((lane >> 4) << 3);
#pragma unroll
        for (int tn = 0; tn < 4; ++tn) {
            const int colS = tn * 16 + (lane & 15);
#pragma unroll
            for (int v = 0; v < 8; ++v) {
                float p = e[tn][v] * __builtin_amdgcn_rcpf(psum[v]);
                Pt[colS * STRIDE + (row0 + v)] = (bf16_t)p;  // P^T[m][w]
            }
        }
    }

    // ================= GEMM4: out = X @ P  (stage f32 tile in LDS) =================
#pragma unroll
    for (int tm = 0; tm < 4; ++tm)
#pragma unroll
        for (int tn = 0; tn < 4; ++tn) {
            v8f acc = gemm_tile64(Xb, Pt, tm * 16, tn * 16, lane);
            const int colO = tn * 16 + (lane & 15);
            const int row0 = tm * 16 + ((lane >> 4) << 3);
#pragma unroll
            for (int v = 0; v < 8; ++v)
                Ob[(row0 + v) * OSTRIDE + colO] = acc[v];
        }
    __syncthreads();   // staged outputs are read cross-wave

    // ---- cooperative, coalesced store: out[b,h,m,c0..c0+3] in 16 B groups ----
    {
        float* op = out + (size_t)b * 64 * 64 * 256 + c0;
        const int co = tid & 3;
        const int p0 = tid >> 2;
        const float* Oc = reinterpret_cast<const float*>(
            smem_raw + co * REGION_BYTES + OUT_BYTE_OFF + co * 32);
        for (int hw = p0; hw < 4096; hw += 32) {
            op[(size_t)hw * 256 + co] = Oc[(hw >> 6) * OSTRIDE + (hw & 63)];
        }
    }
}

extern "C" void kernel_launch(void* const* d_in, const int* in_sizes, int n_in,
                              void* d_out, int out_size, void* d_ws, size_t ws_size,
                              hipStream_t stream) {
    (void)in_sizes; (void)n_in; (void)d_ws; (void)ws_size; (void)out_size;
    const float* x  = (const float*)d_in[0];
    const float* Wq = (const float*)d_in[1];
    const float* Wk = (const float*)d_in[2];
    float* out      = (float*)d_out;

    const size_t shmem = 4u * (size_t)REGION_BYTES;  // 147456 B
    hipFuncSetAttribute((const void*)self_attn_bf16_wmma,
                        hipFuncAttributeMaxDynamicSharedMemorySize, (int)shmem);
    self_attn_bf16_wmma<<<dim3(32 * 64), dim3(128), shmem, stream>>>(x, Wq, Wk, out);
}
